// AttnAdapter_74732430950767
// MI455X (gfx1250) — matrix operand order repaired and checked by reading, not compiled, not run on hardware
//
#include <hip/hip_runtime.h>

#define DEV __device__ __forceinline__

typedef __attribute__((ext_vector_type(16))) __bf16 v16bf;
typedef __attribute__((ext_vector_type(8)))  float  v8f;

struct __attribute__((aligned(16))) B128v { unsigned int w[4]; };

// ---------------- async global->LDS staging (CDNA5 ASYNCcnt path) ----------------
#if defined(__has_builtin)
#if __has_builtin(__builtin_amdgcn_global_load_async_to_lds_b128)
#define HAVE_ASYNC_LDS 1
#endif
#endif
#ifndef HAVE_ASYNC_LDS
#define HAVE_ASYNC_LDS 0
#endif

#if HAVE_ASYNC_LDS
// match the builtin's parameter types exactly:
//   (int __vector(4) addrspace(1)*, int __vector(4) addrspace(3)*, Imm, Imm)
typedef int gvec4 __attribute__((vector_size(16)));
typedef gvec4 __attribute__((address_space(1)))* gv4_g;
typedef gvec4 __attribute__((address_space(3)))* gv4_l;
#endif

// copy 16 bytes (8 bf16) global -> LDS
DEV void stage_b128(const unsigned short* g, unsigned short* l) {
#if HAVE_ASYNC_LDS
  __builtin_amdgcn_global_load_async_to_lds_b128((gv4_g)g, (gv4_l)l, 0, 0);
#else
  *(B128v*)l = *(const B128v*)g;
#endif
}
DEV void async_wait0() {
#if HAVE_ASYNC_LDS
#if __has_builtin(__builtin_amdgcn_s_wait_asynccnt)
  __builtin_amdgcn_s_wait_asynccnt(0);
#else
  asm volatile("s_wait_asynccnt 0x0" ::: "memory");
#endif
#endif
}

// ---------------- bf16 helpers (RNE) ----------------
DEV unsigned short f32_to_bf16(float f) {
  unsigned int u = __builtin_bit_cast(unsigned int, f);
  u += 0x7FFFu + ((u >> 16) & 1u);
  return (unsigned short)(u >> 16);
}
DEV float bf16_to_f32(unsigned short h) {
  unsigned int u = ((unsigned int)h) << 16;
  return __builtin_bit_cast(float, u);
}
DEV v8f zero8() { v8f z = {0.f,0.f,0.f,0.f,0.f,0.f,0.f,0.f}; return z; }

// ---------------- WMMA fragment loaders (CDNA5 wave32 layouts) ----------------
// A fragment 16x32 bf16: lane m(0..15), hi=lane>>4.
//   element e -> K = 16*hi + e
DEV v16bf load_A_frag(const unsigned short* base, int stride, int lane, int kbase) {
  int m = lane & 15, hi = lane >> 4;
  const unsigned short* p = base + m * stride + kbase;
  union { v16bf v; B128v q[2]; } u;
  u.q[0] = *(const B128v*)(p + 16 * hi);
  u.q[1] = *(const B128v*)(p + 16 * hi + 8);
  return u.v;
}
// B fragment 32x16 bf16: lane holds column n = lane&15, K = e + 16*(lane>>4)
DEV v16bf load_B_frag(const unsigned short* base, int stride, int lane) {
  int n = lane & 15, hi = lane >> 4;
  const unsigned short* p = base + n * stride + 16 * hi;
  union { v16bf v; B128v q[2]; } u;
  u.q[0] = *(const B128v*)(p);
  u.q[1] = *(const B128v*)(p + 8);
  return u.v;
}

DEV v8f wmma_bf16(v16bf a, v16bf b, v8f c) {
  return __builtin_amdgcn_wmma_f32_16x16x32_bf16(false, a, false, b, (short)0, c, false, false);
}

// row reductions within the 16-lane half-groups of a wave32
DEV float redmax16(float x) {
  #pragma unroll
  for (int m = 1; m < 16; m <<= 1) x = fmaxf(x, __shfl_xor(x, m, 32));
  return x;
}
DEV float redsum16(float x) {
  #pragma unroll
  for (int m = 1; m < 16; m <<= 1) x += __shfl_xor(x, m, 32);
  return x;
}

// ---------------- fp32 -> bf16 conversion (vectorized: 4 floats/thread/iter) ----------------
__global__ void cvt_bf16_kernel(const float* __restrict__ in,
                                unsigned short* __restrict__ out, size_t n4) {
  size_t stride = (size_t)gridDim.x * blockDim.x;
  for (size_t i = (size_t)blockIdx.x * blockDim.x + threadIdx.x; i < n4; i += stride) {
    float4 f = ((const float4*)in)[i];
    unsigned int lo = (unsigned int)f32_to_bf16(f.x) | ((unsigned int)f32_to_bf16(f.y) << 16);
    unsigned int hi = (unsigned int)f32_to_bf16(f.z) | ((unsigned int)f32_to_bf16(f.w) << 16);
    ((uint2*)out)[i] = make_uint2(lo, hi);
  }
}

// ---------------- tiled bf16 GEMM: C[M,N] = A[M,K] @ B[N,K]^T ----------------
// 256 threads = 8 waves in a 4(M)x2(N) grid; tile 128x128, K-step 32,
// double-buffered LDS fed by async global->LDS loads.
template<bool WRITE_F32>
__global__ __launch_bounds__(256)
void gemm_bf16_kernel(const unsigned short* __restrict__ A,
                      const unsigned short* __restrict__ B,
                      void* __restrict__ Cout, int M, int N, int K) {
  __shared__ __attribute__((aligned(16))) unsigned short ldsA[2][128 * 32];
  __shared__ __attribute__((aligned(16))) unsigned short ldsB[2][128 * 32];
  const int tid = threadIdx.x;
  const int wave = tid >> 5, lane = tid & 31;
  const int hi = lane >> 4, ln = lane & 15;
  const int wm = wave >> 1, wn = wave & 1;
  const int m0 = blockIdx.y * 128, n0 = blockIdx.x * 128;
  const int arow = tid >> 1, acol = (tid & 1) * 16; // 2 threads/row, 16 halfs each

  v8f acc[2][4];
  #pragma unroll
  for (int st = 0; st < 2; ++st)
    #pragma unroll
    for (int j = 0; j < 4; ++j) acc[st][j] = zero8();

  const int NS = K / 32;
  // prologue: stage k-step 0 into buffer 0
  {
    const unsigned short* ga = A + (size_t)(m0 + arow) * K + acol;
    const unsigned short* gb = B + (size_t)(n0 + arow) * K + acol;
    unsigned short* la = &ldsA[0][arow * 32 + acol];
    unsigned short* lb = &ldsB[0][arow * 32 + acol];
    stage_b128(ga, la); stage_b128(ga + 8, la + 8);
    stage_b128(gb, lb); stage_b128(gb + 8, lb + 8);
  }
  for (int s = 0; s < NS; ++s) {
    const int cur = s & 1;
    async_wait0();       // my async loads into buf[cur] have landed
    __syncthreads();     // everyone's landed; previous compute on buf[cur^1] done
    if (s + 1 < NS) {    // prefetch next K-step into the other buffer
      const int k0 = (s + 1) * 32;
      const unsigned short* ga = A + (size_t)(m0 + arow) * K + k0 + acol;
      const unsigned short* gb = B + (size_t)(n0 + arow) * K + k0 + acol;
      unsigned short* la = &ldsA[cur ^ 1][arow * 32 + acol];
      unsigned short* lb = &ldsB[cur ^ 1][arow * 32 + acol];
      stage_b128(ga, la); stage_b128(ga + 8, la + 8);
      stage_b128(gb, lb); stage_b128(gb + 8, lb + 8);
    }
    // compute: 2 A-frags x 4 B-frags -> 8 WMMA
    v16bf a0 = load_A_frag(&ldsA[cur][(wm * 32) * 32], 32, lane, 0);
    v16bf a1 = load_A_frag(&ldsA[cur][(wm * 32 + 16) * 32], 32, lane, 0);
    #pragma unroll
    for (int j = 0; j < 4; ++j) {
      v16bf bf = load_B_frag(&ldsB[cur][(wn * 64 + j * 16) * 32], 32, lane);
      acc[0][j] = wmma_bf16(a0, bf, acc[0][j]);
      acc[1][j] = wmma_bf16(a1, bf, acc[1][j]);
    }
  }

  // C layout: VGPR r -> row r + 8*hi, column = lane&15 within 16-col group
  #pragma unroll
  for (int st = 0; st < 2; ++st) {
    #pragma unroll
    for (int j = 0; j < 4; ++j) {
      int col = n0 + wn * 64 + j * 16 + ln;
      #pragma unroll
      for (int r = 0; r < 8; ++r) {
        int row = m0 + wm * 32 + st * 16 + r + 8 * hi;
        if constexpr (WRITE_F32)
          ((float*)Cout)[(size_t)row * N + col] = acc[st][j][r];
        else
          ((unsigned short*)Cout)[(size_t)row * N + col] = f32_to_bf16(acc[st][j][r]);
      }
    }
  }
}

// ---------------- RoPE (in-place on bf16, Llama-style duplicated half-freqs) ----------------
__global__ void rope_kernel(unsigned short* __restrict__ X,
                            const float* __restrict__ cosb,
                            const float* __restrict__ sinb,
                            int S, int nH) {
  int total = S * nH * 64;
  int stride = gridDim.x * blockDim.x;
  for (int idx = blockIdx.x * blockDim.x + threadIdx.x; idx < total; idx += stride) {
    int d = idx & 63;
    int t = idx >> 6;
    int hh = t % nH;
    int s  = t / nH;
    size_t base = (size_t)s * ((size_t)nH * 128) + (size_t)hh * 128;
    float q0 = bf16_to_f32(X[base + d]);
    float q1 = bf16_to_f32(X[base + d + 64]);
    float c0 = cosb[s * 128 + d],      s0 = sinb[s * 128 + d];
    float c1 = cosb[s * 128 + d + 64], s1 = sinb[s * 128 + d + 64];
    X[base + d]      = f32_to_bf16(q0 * c0 - q1 * s0);
    X[base + d + 64] = f32_to_bf16(q1 * c1 + q0 * s1);
  }
}

// ---------------- flash attention with logit biases, K/V shared across GQA group ----------------
// grid = (S/32, KV); 256 threads = 8 waves = 4 query heads x 2 query strips of 16 rows.
// K/V for one kv head staged once per block and reused by all 4 query heads.
__global__ __launch_bounds__(256)
void flash_attn_kernel(const unsigned short* __restrict__ Qb,
                       const unsigned short* __restrict__ Kb,
                       const unsigned short* __restrict__ Vb,
                       const float* __restrict__ salience,
                       unsigned short* __restrict__ Ob,
                       const int* __restrict__ img_start_p,
                       const int* __restrict__ img_end_p,
                       const int* __restrict__ sys_end_p) {
  const int D = 128, HD = 4096, KVD = 1024;
  __shared__ __attribute__((aligned(16))) unsigned short ldsQ[4 * 32 * 128]; // [head_l][qrow][d]
  __shared__ __attribute__((aligned(16))) unsigned short ldsK[64 * 128];     // [key][d]
  __shared__ __attribute__((aligned(16))) unsigned short ldsVT[128 * 64];    // [d][key]
  __shared__ __attribute__((aligned(16))) unsigned short ldsP[8 * 16 * 64];  // per-wave P

  const int tid = threadIdx.x, wave = tid >> 5, lane = tid & 31;
  const int hi = lane >> 4, ln = lane & 15;
  const int head_l = wave >> 1;           // 0..3 query head within GQA group
  const int strip  = wave & 1;            // 0..1 strip of 16 query rows
  const int s0  = blockIdx.x * 32;
  const int kvh = blockIdx.y;
  const int h   = kvh * 4 + head_l;
  const int img_start = img_start_p[0];
  const int img_end   = img_end_p[0];
  const int sys_end   = sys_end_p[0];
  const float scale = 0.08838834764831845f; // 1/sqrt(128)

  { // stage Q: 4 heads x 32 rows x 128 d (async)
    int combo = tid >> 1, half = tid & 1;
    int hl = combo >> 5, qr = combo & 31;
    const unsigned short* src =
        Qb + (size_t)(s0 + qr) * HD + (kvh * 4 + hl) * D + half * 64;
    unsigned short* dst = &ldsQ[(hl * 32 + qr) * 128 + half * 64];
    #pragma unroll
    for (int i = 0; i < 8; ++i) stage_b128(src + i * 8, dst + i * 8);
  }

  float m_run[8], l_run[8];
  v8f o[8];
  #pragma unroll
  for (int r = 0; r < 8; ++r) { m_run[r] = -3.0e38f; l_run[r] = 0.f; }
  #pragma unroll
  for (int jj = 0; jj < 8; ++jj) o[jj] = zero8();

  const int kb_end = blockIdx.x >> 1; // last key block touching q rows s0..s0+31
  for (int kb = 0; kb <= kb_end; ++kb) {
    const int key0 = kb * 64;
    __syncthreads(); // previous block's compute on ldsK/ldsVT is done
    { // stage K block 64x128 (async)
      int krow = tid >> 2, kq = tid & 3;
      const unsigned short* src = Kb + (size_t)(key0 + krow) * KVD + kvh * D + kq * 32;
      unsigned short* dst = &ldsK[krow * 128 + kq * 32];
      #pragma unroll
      for (int i = 0; i < 4; ++i) stage_b128(src + i * 8, dst + i * 8);
    }
    { // stage V transposed: key-pairs packed into aligned b32 LDS stores
      int kp = tid & 31;         // keys 2kp, 2kp+1
      int dc = tid >> 5;         // d chunk of 16
      const unsigned short* va = Vb + (size_t)(key0 + 2 * kp) * KVD + kvh * D + dc * 16;
      const unsigned short* vb = Vb + (size_t)(key0 + 2 * kp + 1) * KVD + kvh * D + dc * 16;
      unsigned short aL[16], bL[16];
      *(B128v*)&aL[0] = ((const B128v*)va)[0];
      *(B128v*)&aL[8] = ((const B128v*)va)[1];
      *(B128v*)&bL[0] = ((const B128v*)vb)[0];
      *(B128v*)&bL[8] = ((const B128v*)vb)[1];
      #pragma unroll
      for (int e = 0; e < 16; ++e) {
        int d = dc * 16 + e;
        unsigned int pack = (unsigned int)aL[e] | ((unsigned int)bL[e] << 16);
        *(unsigned int*)&ldsVT[d * 64 + 2 * kp] = pack;
      }
    }
    async_wait0();
    __syncthreads();

    // S = Q K^T (16 q rows x 64 keys per wave)
    v8f sacc[4];
    #pragma unroll
    for (int j = 0; j < 4; ++j) sacc[j] = zero8();
    const unsigned short* qbase = &ldsQ[(head_l * 32 + strip * 16) * 128];
    #pragma unroll
    for (int d0 = 0; d0 < 128; d0 += 32) {
      v16bf af = load_A_frag(qbase, 128, lane, d0);
      #pragma unroll
      for (int j = 0; j < 4; ++j) {
        v16bf bf = load_B_frag(&ldsK[(j * 16) * 128 + d0], 128, lane);
        sacc[j] = wmma_bf16(af, bf, sacc[j]);
      }
    }

    // per-key logit bias (system-prompt suppression + salience bias on image keys)
    float bias[4];
    #pragma unroll
    for (int j = 0; j < 4; ++j) {
      int key = key0 + j * 16 + ln;
      float b = 0.f;
      if (key <= sys_end) b -= 0.1f;
      if (key >= img_start && key <= img_end) {
        float sal = salience[key - img_start];
        b += 1.0f * sal - 0.1f * (1.f - sal);
      }
      bias[j] = b;
    }

    // online softmax, C-layout rows r + 8*hi
    #pragma unroll
    for (int r = 0; r < 8; ++r) {
      int qrow = s0 + strip * 16 + r + 8 * hi;
      float sv[4];
      float mx = -3.0e38f;
      #pragma unroll
      for (int j = 0; j < 4; ++j) {
        int key = key0 + j * 16 + ln;
        float x = sacc[j][r] * scale + bias[j];
        if (key > qrow) x = -1.0e9f; // causal
        sv[j] = x;
        mx = fmaxf(mx, x);
      }
      mx = redmax16(mx);
      float m_new = fmaxf(m_run[r], mx);
      float alpha = __expf(m_run[r] - m_new);
      float ps = 0.f;
      #pragma unroll
      for (int j = 0; j < 4; ++j) {
        float p = __expf(sv[j] - m_new);
        ps += p;
        ldsP[wave * 1024 + (r + 8 * hi) * 64 + j * 16 + ln] = f32_to_bf16(p);
      }
      ps = redsum16(ps);
      l_run[r] = l_run[r] * alpha + ps;
      m_run[r] = m_new;
      #pragma unroll
      for (int jj = 0; jj < 8; ++jj) o[jj][r] *= alpha;
    }
    // ldsP is per-wave; LDS ops are in-order within a wave -> no barrier needed

    // O += P V  (K-dim = 64 keys => two 16x32 A fragments)
    #pragma unroll
    for (int kk = 0; kk < 64; kk += 32) {
      v16bf pf = load_A_frag(&ldsP[wave * 1024], 64, lane, kk);
      #pragma unroll
      for (int jj = 0; jj < 8; ++jj) {
        v16bf vf = load_B_frag(&ldsVT[(jj * 16) * 64 + kk], 64, lane);
        o[jj] = wmma_bf16(pf, vf, o[jj]);
      }
    }
  }

  // epilogue: normalize and store bf16 O
  #pragma unroll
  for (int jj = 0; jj < 8; ++jj) {
    int d = jj * 16 + ln;
    #pragma unroll
    for (int r = 0; r < 8; ++r) {
      int row = s0 + strip * 16 + r + 8 * hi;
      Ob[(size_t)row * HD + h * D + d] = f32_to_bf16(o[jj][r] / l_run[r]);
    }
  }
}

// ---------------- launcher ----------------
extern "C" void kernel_launch(void* const* d_in, const int* in_sizes, int n_in,
                              void* d_out, int out_size, void* d_ws, size_t ws_size,
                              hipStream_t stream) {
  const size_t S = 2048, HID = 4096, HD = 4096, KVD = 1024;
  const float* hs   = (const float*)d_in[0];
  const float* cosb = (const float*)d_in[1];
  const float* sinb = (const float*)d_in[2];
  // d_in[3] = attention_mask (pure causal; applied analytically)
  const float* sal  = (const float*)d_in[4];
  const float* Wq   = (const float*)d_in[5];
  const float* Wk   = (const float*)d_in[6];
  const float* Wv   = (const float*)d_in[7];
  const float* Wo   = (const float*)d_in[8];
  const int* isp = (const int*)d_in[9];
  const int* iep = (const int*)d_in[10];
  const int* sep = (const int*)d_in[11];

  unsigned short* Xb  = (unsigned short*)d_ws;       size_t off = S * HID;
  unsigned short* Wqb = Xb + off;                    off += HD * HID;
  unsigned short* Wkb = Xb + off;                    off += KVD * HID;
  unsigned short* Wvb = Xb + off;                    off += KVD * HID;
  unsigned short* Wob = Xb + off;                    off += HID * HD;
  unsigned short* Qb  = Xb + off;                    off += S * HD;
  unsigned short* Kb2 = Xb + off;                    off += S * KVD;
  unsigned short* Vb2 = Xb + off;                    off += S * KVD;
  unsigned short* Ob  = Xb + off;                    off += S * HD;

  // fp32 -> bf16 staging (sizes all divisible by 4)
  cvt_bf16_kernel<<<1024, 256, 0, stream>>>(hs, Xb,  (S * HID) / 4);
  cvt_bf16_kernel<<<1024, 256, 0, stream>>>(Wq, Wqb, (HD * HID) / 4);
  cvt_bf16_kernel<<<1024, 256, 0, stream>>>(Wk, Wkb, (KVD * HID) / 4);
  cvt_bf16_kernel<<<1024, 256, 0, stream>>>(Wv, Wvb, (KVD * HID) / 4);
  cvt_bf16_kernel<<<1024, 256, 0, stream>>>(Wo, Wob, (HID * HD) / 4);

  // projections (tile 128x128)
  gemm_bf16_kernel<false><<<dim3(HD / 128,  S / 128), 256, 0, stream>>>(Xb, Wqb, Qb,  (int)S, (int)HD,  (int)HID);
  gemm_bf16_kernel<false><<<dim3(KVD / 128, S / 128), 256, 0, stream>>>(Xb, Wkb, Kb2, (int)S, (int)KVD, (int)HID);
  gemm_bf16_kernel<false><<<dim3(KVD / 128, S / 128), 256, 0, stream>>>(Xb, Wvb, Vb2, (int)S, (int)KVD, (int)HID);

  // RoPE (in-place; Q/K freshly rewritten by the GEMMs every call)
  rope_kernel<<<512, 256, 0, stream>>>(Qb,  cosb, sinb, (int)S, 32);
  rope_kernel<<<512, 256, 0, stream>>>(Kb2, cosb, sinb, (int)S, 8);

  // streaming attention, K/V shared across each GQA group
  flash_attn_kernel<<<dim3(S / 32, 8), 256, 0, stream>>>(Qb, Kb2, Vb2, sal, Ob, isp, iep, sep);

  // output projection -> fp32 d_out
  gemm_bf16_kernel<true><<<dim3(HID / 128, S / 128), 256, 0, stream>>>(Ob, Wob, d_out, (int)S, (int)HID, (int)HD);
}
